// SimpleRNN_25262997635797
// MI455X (gfx1250) — matrix operand order repaired and checked by reading, not compiled, not run on hardware
//
#include <hip/hip_runtime.h>
#include <hip/hip_bf16.h>

typedef __bf16 bf16_t;
typedef __attribute__((ext_vector_type(16))) __bf16 v16bf;
typedef __attribute__((ext_vector_type(8)))  __bf16 v8bf;
typedef __attribute__((ext_vector_type(8)))  float  v8f;
typedef __attribute__((ext_vector_type(4)))  int    v4i;

#define RNN_B 64
#define RNN_T 512
#define RNN_I 256
#define RNN_H 1024
#define RNN_O 1000

// ---- CDNA5 async global->LDS path (guarded; falls back to sync copy) ------
#if defined(__has_builtin)
#if __has_builtin(__builtin_amdgcn_global_load_async_to_lds_b128)
#define HAVE_ASYNC_LDS 1
#endif
#endif

#ifdef HAVE_ASYNC_LDS
typedef __attribute__((address_space(1))) v4i as1_v4i;
typedef __attribute__((address_space(3))) v4i as3_v4i;
__device__ __forceinline__ void async_copy16(const void* gsrc, void* ldst) {
  // integer round-trips avoid generic<->AS pointer casts; AS3 ptr = LDS offset
  as1_v4i* g = (as1_v4i*)(unsigned long long)gsrc;
  as3_v4i* l = (as3_v4i*)(unsigned int)(unsigned long long)ldst;
  __builtin_amdgcn_global_load_async_to_lds_b128(g, l, 0, 0);
}
__device__ __forceinline__ void async_wait0() {
#if __has_builtin(__builtin_amdgcn_s_wait_asynccnt)
  __builtin_amdgcn_s_wait_asynccnt(0);
#else
  asm volatile("s_wait_asynccnt 0x0" ::: "memory");
#endif
}
#endif

// ---------------------------------------------------------------------------
// fp32 -> bf16 conversion (weights reused every step; convert once)
// ---------------------------------------------------------------------------
__global__ void cvt_f32_to_bf16(const float* __restrict__ in,
                                bf16_t* __restrict__ out, long n) {
  long i = (long)blockIdx.x * blockDim.x + threadIdx.x;
  long stride = (long)gridDim.x * blockDim.x;
  for (; i < n; i += stride) out[i] = (bf16_t)in[i];
}

// ---------------------------------------------------------------------------
// WMMA fragment helpers (bf16 16x16x32, this code's contiguous k order, same for A and B)
// A 16x32: lane covers row M=lane&15, K half = (lane>>4)*16, contiguous
// B 32x16: lane covers column N=lane&15, K half = (lane>>4)*16, contiguous
// C/D:     VGPR r -> M = (lane>>4)*8 + r, N = lane&15
// ---------------------------------------------------------------------------
__device__ __forceinline__ v16bf make_A_frag(const bf16_t* p /* row base + kh*16 */) {
  v8bf lo = *(const v8bf*)p;
  v8bf hi = *(const v8bf*)(p + 8);
  v16bf a;
#pragma unroll
  for (int i = 0; i < 8; ++i) { a[i] = lo[i]; a[8 + i] = hi[i]; }
  return a;
}

// ---------------------------------------------------------------------------
// C(M,N) = A(M,K)bf16 @ W(N,K)bf16^T + bias, fp32 accumulate/output.
// Block = 256 threads = 8 waves -> 32(M) x 64(N) tile; wave = one 16x16 tile.
// swizzleT != 0 scatters output row m=b*T+t to row t*B+b ((T,B,H) layout).
// ---------------------------------------------------------------------------
__global__ void gemm_bias_wmma(const bf16_t* __restrict__ A,
                               const bf16_t* __restrict__ W,
                               const float* __restrict__ bias,
                               float* __restrict__ C,
                               int M, int N, int K, int swizzleT) {
  int wave = threadIdx.x >> 5;
  int lane = threadIdx.x & 31;
  int m0 = blockIdx.y * 32 + (wave & 1) * 16;
  int n0 = blockIdx.x * 64 + (wave >> 1) * 16;

  int ln = lane & 15;
  int lh = lane >> 4;
  int mrow = m0 + ln;
  int ncol = n0 + ln;
  int ncl = ncol < N ? ncol : N - 1;   // clamp for ragged N (logits N=1000)

  const bf16_t* Ap = A + (long)mrow * K + lh * 16;
  const bf16_t* Wp = W + (long)ncl * K + lh * 16;

  v8f acc;
  float bv = bias[ncl];
#pragma unroll
  for (int r = 0; r < 8; ++r) acc[r] = bv;

  for (int k = 0; k < K; k += 32) {
    v16bf a = make_A_frag(Ap + k);
    v16bf b = *(const v16bf*)(Wp + k);
    acc = __builtin_amdgcn_wmma_f32_16x16x32_bf16(false, a, false, b,
                                                  (short)0, acc, false, false);
  }

  if (ncol < N) {
    int mbase = lh * 8;
#pragma unroll
    for (int r = 0; r < 8; ++r) {
      int m = m0 + mbase + r;
      long row = swizzleT ? ((long)(m % swizzleT) * (M / swizzleT) + (m / swizzleT))
                          : (long)m;
      C[row * N + ncol] = acc[r];
    }
  }
}

// ---------------------------------------------------------------------------
// scan init: zero h ping-pong buffers + grid-sync counter
// ---------------------------------------------------------------------------
__global__ void scan_init(bf16_t* __restrict__ hbuf, unsigned* __restrict__ counter,
                          int n) {
  int i = blockIdx.x * blockDim.x + threadIdx.x;
  if (i == 0) *counter = 0u;
  for (; i < n; i += gridDim.x * blockDim.x) hbuf[i] = (bf16_t)0.0f;
}

// ---------------------------------------------------------------------------
// Persistent recurrent scan: h_t = relu(pre[t] + h_{t-1} @ Whh^T + bhh)
// pre: (T,B,H) fp32.  Whh: (H,H) bf16.  hping: 2*B*H bf16 ping-pong.
// hs_out: (B*T,H) bf16 or null.
// Grid = 64 blocks x 256 thr (8 waves): 4 N-tiles x 2-way K-split per block.
// A block (16 x H of h_prev) staged to LDS via async-to-LDS; K-halves reduced
// through LDS. Grid-wide step barrier via fence + atomic counter.
// ---------------------------------------------------------------------------
__global__ void __launch_bounds__(256)
rnn_scan_wmma(const float* __restrict__ pre,
              const bf16_t* __restrict__ Whh,
              const float* __restrict__ bhh,
              bf16_t* __restrict__ hping,
              bf16_t* __restrict__ hs_out,
              unsigned* __restrict__ counter,
              int T) {
  const int H = RNN_H, Bb = RNN_B;
  const int LDA = RNN_H + 8;                 // padded LDS stride (bf16 elems)
  __shared__ bf16_t Ash[16 * (RNN_H + 8)];
  __shared__ v8f Red[4 * 32];                // partial sums from K-half 1

  int wave = threadIdx.x >> 5;
  int lane = threadIdx.x & 31;
  int ntile = wave & 3;                      // 4 N-tiles per block
  int kh2 = wave >> 2;                       // K-half: 0 or 1
  int m0 = (blockIdx.x >> 4) * 16;           // 0,16,32,48
  int n0 = (blockIdx.x & 15) * 64 + ntile * 16;
  int ln = lane & 15;
  int lh = lane >> 4;
  int mbase = lh * 8;

  float bv = bhh[n0 + ln];
  const bf16_t* Wp = Whh + (long)(n0 + ln) * H + kh2 * (H / 2) + lh * 16;

  for (int t = 0; t < T; ++t) {
    const bf16_t* hprev = hping + (long)(t & 1) * Bb * H;
    bf16_t* hcur = hping + (long)((t + 1) & 1) * Bb * H;

    // stage 16 rows x H of h_prev into LDS (16B chunks, 256 threads)
    for (int idx = threadIdx.x; idx < 16 * (RNN_H / 8); idx += blockDim.x) {
      int r = idx >> 7;                      // H/8 = 128 chunks per row
      int c = idx & 127;
      const bf16_t* src = hprev + (long)(m0 + r) * H + c * 8;
      bf16_t* dst = &Ash[r * LDA + c * 8];
#ifdef HAVE_ASYNC_LDS
      async_copy16(src, dst);
#else
      *(uint4*)dst = *(const uint4*)src;
#endif
    }
#ifdef HAVE_ASYNC_LDS
    async_wait0();
#endif
    __syncthreads();

    v8f acc;
    if (kh2 == 0) {
      const float* pre_t = pre + (long)t * Bb * H;
#pragma unroll
      for (int r = 0; r < 8; ++r)
        acc[r] = bv + pre_t[(long)(m0 + mbase + r) * H + (n0 + ln)];
    } else {
#pragma unroll
      for (int r = 0; r < 8; ++r) acc[r] = 0.0f;
    }

    const bf16_t* arow = &Ash[ln * LDA + kh2 * (H / 2)];
    for (int k = 0; k < H / 2; k += 32) {
      if (k + 256 <= H / 2) __builtin_prefetch(Wp + k + 224, 0, 0);
      v16bf a = make_A_frag(arow + k + lh * 16);
      v16bf b = *(const v16bf*)(Wp + k);
      acc = __builtin_amdgcn_wmma_f32_16x16x32_bf16(false, a, false, b,
                                                    (short)0, acc, false, false);
    }

    // reduce the two K-halves through LDS; K-half-0 waves do relu + store
    if (kh2 == 1) Red[ntile * 32 + lane] = acc;
    __syncthreads();
    if (kh2 == 0) {
      v8f other = Red[ntile * 32 + lane];
#pragma unroll
      for (int r = 0; r < 8; ++r) {
        float v = acc[r] + other[r];
        v = v > 0.0f ? v : 0.0f;
        bf16_t hv = (bf16_t)v;
        int m = m0 + mbase + r;
        hcur[(long)m * H + (n0 + ln)] = hv;
        if (hs_out) hs_out[((long)m * T + t) * H + (n0 + ln)] = hv;
      }
    }

    // ---- grid-wide step barrier ----
    __threadfence();
    __syncthreads();
    if (threadIdx.x == 0) {
      atomicAdd(counter, 1u);
      unsigned target = (unsigned)gridDim.x * (unsigned)(t + 1);
      while (*(volatile unsigned*)counter < target) __builtin_amdgcn_s_sleep(2);
    }
    __syncthreads();
  }
}

// ---------------------------------------------------------------------------
// row-wise log-softmax: one block per batch row
// ---------------------------------------------------------------------------
__global__ void log_softmax_rows(const float* __restrict__ logits,
                                 float* __restrict__ out, int N) {
  __shared__ float red[256];
  int row = blockIdx.x;
  const float* x = logits + (long)row * N;

  float m = -__builtin_inff();
  for (int j = threadIdx.x; j < N; j += blockDim.x) m = fmaxf(m, x[j]);
  red[threadIdx.x] = m;
  __syncthreads();
  for (int s = blockDim.x >> 1; s > 0; s >>= 1) {
    if ((int)threadIdx.x < s) red[threadIdx.x] = fmaxf(red[threadIdx.x], red[threadIdx.x + s]);
    __syncthreads();
  }
  m = red[0];
  __syncthreads();

  float sum = 0.0f;
  for (int j = threadIdx.x; j < N; j += blockDim.x) sum += __expf(x[j] - m);
  red[threadIdx.x] = sum;
  __syncthreads();
  for (int s = blockDim.x >> 1; s > 0; s >>= 1) {
    if ((int)threadIdx.x < s) red[threadIdx.x] += red[threadIdx.x + s];
    __syncthreads();
  }
  float lse = m + __logf(red[0]);
  for (int j = threadIdx.x; j < N; j += blockDim.x)
    out[(long)row * N + j] = x[j] - lse;
}

// ---------------------------------------------------------------------------
extern "C" void kernel_launch(void* const* d_in, const int* in_sizes, int n_in,
                              void* d_out, int out_size, void* d_ws, size_t ws_size,
                              hipStream_t stream) {
  const int B = RNN_B, T = RNN_T, I = RNN_I, H = RNN_H, O = RNN_O;

  const float* x     = (const float*)d_in[0];
  const float* w_ih0 = (const float*)d_in[1];
  const float* w_hh0 = (const float*)d_in[2];
  const float* b_ih0 = (const float*)d_in[3];
  const float* b_hh0 = (const float*)d_in[4];
  const float* w_ih1 = (const float*)d_in[5];
  const float* w_hh1 = (const float*)d_in[6];
  const float* b_ih1 = (const float*)d_in[7];
  const float* b_hh1 = (const float*)d_in[8];
  const float* w_lin = (const float*)d_in[9];
  const float* b_lin = (const float*)d_in[10];

  char* ws = (char*)d_ws;
  size_t off = 0;
  auto take = [&](size_t bytes) -> char* {
    char* p = ws + off;
    off = (off + bytes + 255) & ~(size_t)255;
    return p;
  };

  bf16_t* xb    = (bf16_t*)take((size_t)B * T * I * 2);
  bf16_t* wih0b = (bf16_t*)take((size_t)H * I * 2);
  bf16_t* whh0b = (bf16_t*)take((size_t)H * H * 2);
  bf16_t* wih1b = (bf16_t*)take((size_t)H * H * 2);
  bf16_t* whh1b = (bf16_t*)take((size_t)H * H * 2);
  bf16_t* wlinb = (bf16_t*)take((size_t)O * H * 2);
  float*  pre   = (float*)take((size_t)B * T * H * 4);   // reused for both layers
  bf16_t* hs0b  = (bf16_t*)take((size_t)B * T * H * 2);
  bf16_t* hping = (bf16_t*)take((size_t)2 * B * H * 2);
  float*  logit = (float*)take((size_t)B * O * 4);
  unsigned* ctr = (unsigned*)take(256);

  // --- convert inputs/weights to bf16 ---
  auto cvt = [&](const float* src, bf16_t* dst, long n) {
    int blocks = (int)((n + 255) / 256);
    if (blocks > 2048) blocks = 2048;
    cvt_f32_to_bf16<<<blocks, 256, 0, stream>>>(src, dst, n);
  };
  cvt(x,     xb,    (long)B * T * I);
  cvt(w_ih0, wih0b, (long)H * I);
  cvt(w_hh0, whh0b, (long)H * H);
  cvt(w_ih1, wih1b, (long)H * H);
  cvt(w_hh1, whh1b, (long)H * H);
  cvt(w_lin, wlinb, (long)O * H);

  // --- layer 0: pre0 = x @ w_ih0^T + b_ih0, scattered to (T,B,H) ---
  gemm_bias_wmma<<<dim3(H / 64, (B * T) / 32), 256, 0, stream>>>(
      xb, wih0b, b_ih0, pre, B * T, H, I, T);

  scan_init<<<64, 256, 0, stream>>>(hping, ctr, 2 * B * H);
  rnn_scan_wmma<<<64, 256, 0, stream>>>(pre, whh0b, b_hh0, hping, hs0b, ctr, T);

  // --- layer 1: pre1 = hs0 @ w_ih1^T + b_ih1, scattered to (T,B,H) ---
  gemm_bias_wmma<<<dim3(H / 64, (B * T) / 32), 256, 0, stream>>>(
      hs0b, wih1b, b_ih1, pre, B * T, H, H, T);

  scan_init<<<64, 256, 0, stream>>>(hping, ctr, 2 * B * H);
  rnn_scan_wmma<<<64, 256, 0, stream>>>(pre, whh1b, b_hh1, hping, nullptr, ctr, T);

  // final hidden state: t=T-1 wrote buffer index (T & 1) == 0
  bf16_t* hlast = hping + (long)(T & 1) * B * H;

  // --- logits = last @ w_lin^T + b_lin  (N=1000, guarded tiles) ---
  gemm_bias_wmma<<<dim3((O + 63) / 64, B / 32), 256, 0, stream>>>(
      hlast, wlinb, b_lin, logit, B, O, H, 0);

  // --- log_softmax ---
  log_softmax_rows<<<B, 256, 0, stream>>>(logit, (float*)d_out, O);
}
